// Mamba2LM_738734375256
// MI455X (gfx1250) — matrix operand order repaired and checked by reading, not compiled, not run on hardware
//
#include <hip/hip_runtime.h>

// ---------------- model constants (match reference) ----------------
constexpr int DIM     = 1024;
constexpr int DEPTH   = 2;
constexpr int DSTATE  = 128;
constexpr int HEADDIM = 64;
constexpr int DCONV   = 4;
constexpr int DINNER  = 2048;                         // EXPAND*DIM
constexpr int NHEADS  = DINNER / HEADDIM;             // 32
constexpr int CONVCH  = DINNER + 2 * DSTATE;          // 2304
constexpr int DINPROJ = 2 * DINNER + 2 * DSTATE + NHEADS; // 4384
constexpr int VOCAB   = 256;
constexpr float EPS   = 1e-5f;
constexpr int B = 2, T = 1024;
constexpr int BT = B * T;                             // 2048
constexpr int LCH = 32;                               // scan chunk length
constexpr int NCH = T / LCH;                          // 32 chunks

// ---------------- types for WMMA ----------------
typedef __attribute__((ext_vector_type(16))) __bf16          v16bf;
typedef __attribute__((ext_vector_type(8)))  float           v8f;
typedef __attribute__((ext_vector_type(8)))  unsigned short  u16x8;
typedef __attribute__((ext_vector_type(16))) unsigned short  u16x16;
typedef __attribute__((address_space(3)))    unsigned short* lds_u16p;

static __device__ __forceinline__ unsigned short f2bf(float f) {
    unsigned u = __builtin_bit_cast(unsigned, f);
    unsigned r = u + 0x7FFFu + ((u >> 16) & 1u);   // round-to-nearest-even
    return (unsigned short)(r >> 16);
}
static __device__ __forceinline__ float bf2f(unsigned short u) {
    return __builtin_bit_cast(float, (unsigned)u << 16);
}
static __device__ __forceinline__ float silu(float x) {
    return x / (1.0f + __expf(-x));
}

// A-fragment (16x32 bf16): lane l<16 -> row l, K [0..15];
// lane l>=16 -> row l-16, K [16..31]. rowptr = &buf[tileRow + (lane&15)][0].
static __device__ __forceinline__ v16bf load_a_frag(const unsigned short* rowptr,
                                                    int k0, int half) {
    const unsigned short* p = rowptr + k0 + half * 16;
    u16x8 lo = *(const u16x8*)p;
    u16x8 hi = *(const u16x8*)(p + 8);
    u16x16 av = __builtin_shufflevector(lo, hi,
                    0,1,2,3,4,5,6,7,8,9,10,11,12,13,14,15);
    return __builtin_bit_cast(v16bf, av);
}
// B-fragment (32x16 bf16): lane l -> col l&15; K range [0..15]/[16..31] by half.
static __device__ __forceinline__ v16bf load_b_frag(const unsigned short* colptr,
                                                    int k0, int half) {
    u16x16 bv = *(const u16x16*)(colptr + k0 + half * 16);
    return __builtin_bit_cast(v16bf, bv);
}
static __device__ __forceinline__ v8f wmma_bf16(v16bf a, v16bf b, v8f c) {
    return __builtin_amdgcn_wmma_f32_16x16x32_bf16(false, a, false, b,
                                                   (short)0, c, false, false);
}
// Async global -> LDS copy, 16B per lane (ISA §10 / §15.18, ASYNCcnt-tracked).
static __device__ __forceinline__ void async_load_b128(const unsigned short* lds,
                                                       const unsigned short* g) {
    asm volatile("global_load_async_to_lds_b128 %0, %1, off"
                 :: "v"((lds_u16p)(unsigned long long)(size_t)lds), "v"(g)
                 : "memory");
}

// ---------------- 1) embedding gather ----------------
__global__ __launch_bounds__(256) void embed_gather_kernel(
    const int* __restrict__ x, const float* __restrict__ embed, float* __restrict__ h)
{
    size_t i = (size_t)blockIdx.x * 256 + threadIdx.x;   // over BT*DIM
    if (i >= (size_t)BT * DIM) return;
    int d  = (int)(i % DIM);
    int bt = (int)(i / DIM);
    h[i] = embed[(size_t)x[bt] * DIM + d];
}

// ---------------- 2) fp32 -> bf16 conversion ----------------
__global__ __launch_bounds__(256) void cvt_f32_bf16_kernel(
    const float* __restrict__ in, unsigned short* __restrict__ out, size_t n)
{
    size_t i = (size_t)blockIdx.x * 256 + threadIdx.x;
    size_t stride = (size_t)gridDim.x * 256;
    for (; i < n; i += stride) out[i] = f2bf(in[i]);
}

// ---------------- 3) bf16 WMMA GEMM: D[M,N] = A[M,K] * Bw[N,K]^T (+Cres) ----------------
// Block = 64 M-rows x 128 N-cols (8 waves, one 16-col tile each; 4 M-tiles per wave).
// The shared 64-row A panel is staged global->LDS with double-buffered async copies
// (ASYNCcnt); B weights stream global->register per wave. Inactive waves (N-tail)
// clamp loads, skip stores, and still run all waits/barriers (wave-uniform EXEC).
constexpr int GEMM_NT = 8;      // n-tiles per block
__global__ __launch_bounds__(256) void gemm_bf16_wmma_kernel(
    const unsigned short* __restrict__ A,   // M x K, bf16 row-major
    const unsigned short* __restrict__ Bw,  // N x K, bf16 row-major (weights)
    const float* __restrict__ Cres,         // optional M x N residual (may be null)
    float* __restrict__ D,                  // M x N fp32
    int M, int N, int K)
{
    __shared__ unsigned short sA[2][64][64];   // 16 KB, double-buffered A panel

    const int tid  = threadIdx.x;
    const int wave = tid >> 5, lane = tid & 31;
    const int half = lane >> 4, l15 = lane & 15;

    const int tiles_n = N >> 4;
    const int ngroups = (tiles_n + GEMM_NT - 1) / GEMM_NT;
    const int band    = (int)blockIdx.x / ngroups;       // 64-row M band
    const int ngrp    = (int)blockIdx.x - band * ngroups;
    const int tile_n_real = ngrp * GEMM_NT + wave;
    const bool active = (tile_n_real < tiles_n);
    const int tile_n  = active ? tile_n_real : (tiles_n - 1);  // clamp for safe loads
    const int m0 = band << 6;

    // async staging map: 256 lanes x 16B cover 32 rows x 64 bf16; two issues = 64 rows
    const int srow = tid >> 3;            // 0..31
    const int scol = (tid & 7) * 8;       // 0,8,...,56
    const unsigned short* gA = A + (size_t)(m0 + srow) * K + scol;

    const unsigned short* pb = Bw + (size_t)((tile_n << 4) + l15) * K + half * 16;

    v8f acc[4] = {};
    // prologue: stage chunk 0 into buffer 0
    async_load_b128(&sA[0][srow][scol],      gA);
    async_load_b128(&sA[0][srow + 32][scol], gA + (size_t)32 * K);

    const int nkc = K >> 6;               // 64-wide K chunks
    for (int kc = 0; kc < nkc; ++kc) {
        const int k0 = kc << 6;
        if (kc + 1 < nkc) {               // stage next chunk, then wait for current
            const int nb = (kc + 1) & 1;
            async_load_b128(&sA[nb][srow][scol],      gA + k0 + 64);
            async_load_b128(&sA[nb][srow + 32][scol], gA + (size_t)32 * K + k0 + 64);
            asm volatile("s_wait_asynccnt 2" ::: "memory");
        } else {
            asm volatile("s_wait_asynccnt 0" ::: "memory");
        }
        __syncthreads();                  // publish panel to all waves

        const unsigned short (*pan)[64] = sA[kc & 1];
#pragma unroll
        for (int kk = 0; kk < 64; kk += 32) {
            __builtin_prefetch((const void*)(pb + k0 + kk + 128), 0, 1);
            v16bf b = load_b_frag(pb, k0 + kk, 0);          // half folded into pb
#pragma unroll
            for (int m = 0; m < 4; ++m) {
                v16bf a = load_a_frag(&pan[m * 16 + l15][0] + half * 16, kk, 0);
                acc[m] = wmma_bf16(a, b, acc[m]);
            }
        }
        __syncthreads();                  // all reads done before buffer restage
    }

    if (active) {
        const int col = (tile_n << 4) + l15;
#pragma unroll
        for (int m = 0; m < 4; ++m) {
            const int row0 = m0 + m * 16 + half * 8;
#pragma unroll
            for (int r = 0; r < 8; ++r) {
                size_t off = (size_t)(row0 + r) * N + col;
                float v = acc[m][r];
                if (Cres) v += Cres[off];
                D[off] = v;
            }
        }
    }
}

// ---------------- 4) causal depthwise conv (K=4) + SiLU ----------------
__global__ __launch_bounds__(256) void conv_silu_kernel(
    const float* __restrict__ zx, const float* __restrict__ cw,
    const float* __restrict__ cb, float* __restrict__ xbc)
{
    size_t idx = (size_t)blockIdx.x * 256 + threadIdx.x;   // over BT*CONVCH
    if (idx >= (size_t)BT * CONVCH) return;
    int c  = (int)(idx % CONVCH);
    int bt = (int)(idx / CONVCH);
    int t  = bt % T, b = bt / T;
    const float* base = zx + (size_t)b * T * DINPROJ + DINNER + c;
    float acc = cb[c];
#pragma unroll
    for (int k = 0; k < DCONV; ++k) {
        int tt = t + k - (DCONV - 1);
        if (tt >= 0) acc += base[(size_t)tt * DINPROJ] * cw[c * DCONV + k];
    }
    xbc[idx] = silu(acc);
}

// ---------------- 5) dt = softplus(dt_raw + bias); dA = exp(-exp(alog)*dt) ----------------
__global__ __launch_bounds__(256) void dt_act_kernel(
    const float* __restrict__ zx, const float* __restrict__ dtb,
    const float* __restrict__ alog, float* __restrict__ dt, float* __restrict__ dA)
{
    int idx = blockIdx.x * 256 + threadIdx.x;             // over BT*NHEADS
    if (idx >= BT * NHEADS) return;
    int hh = idx % NHEADS, bt = idx / NHEADS;
    float xr = zx[(size_t)bt * DINPROJ + DINNER + CONVCH + hh] + dtb[hh];
    float d  = (xr > 20.0f) ? xr : log1pf(__expf(xr));
    dt[idx] = d;
    dA[idx] = __expf(-__expf(alog[hh]) * d);
}

// ---------------- 6) chunked selective scan (SSD), WMMA-based ----------------
__global__ __launch_bounds__(256) void chunk_scan_kernel(
    const float* __restrict__ xbc, const float* __restrict__ dt,
    const float* __restrict__ dA, float* __restrict__ yscan)
{
    __shared__ unsigned short sC [LCH][DSTATE];        // 8 KB
    __shared__ unsigned short sB [LCH][DSTATE];        // 8 KB
    __shared__ unsigned short sBT[DSTATE][LCH];        // 8 KB
    __shared__ unsigned short sM [LCH][LCH];           // 2 KB
    __shared__ unsigned short sUt  [HEADDIM][LCH];     // 4 KB
    __shared__ unsigned short sUtld[HEADDIM][LCH];     // 4 KB
    __shared__ unsigned short sS [HEADDIM][DSTATE];    // 16 KB (state, bf16)
    __shared__ float sLc[LCH];

    const int b   = blockIdx.x / NHEADS;
    const int hh  = blockIdx.x % NHEADS;
    const int tid = threadIdx.x;
    const int wave = tid >> 5, lane = tid & 31;
    const int half = lane >> 4, l15 = lane & 15;

    for (int e = tid; e < HEADDIM * DSTATE; e += 256)
        sS[e >> 7][e & 127] = 0;
    __syncthreads();

    for (int c = 0; c < NCH; ++c) {
        const int t0 = c * LCH;
        if (tid == 0) {
            float run = 0.0f;
            for (int s = 0; s < LCH; ++s) {
                run += __logf(dA[((b * T) + t0 + s) * NHEADS + hh]);
                sLc[s] = run;
            }
        }
        __syncthreads();
        const float Lc_last = sLc[LCH - 1];

        for (int e = tid; e < LCH * DSTATE; e += 256) {
            int s = e >> 7, n = e & 127;
            const float* row = xbc + (size_t)(b * T + t0 + s) * CONVCH + DINNER;
            unsigned short bb = f2bf(row[n]);
            sB[s][n] = bb;
            sBT[n][s] = bb;
            sC[s][n] = f2bf(row[DSTATE + n]);
        }
        for (int e = tid; e < HEADDIM * LCH; e += 256) {
            int p = e >> 5, s = e & 31;
            float u = dt[((b * T) + t0 + s) * NHEADS + hh] *
                      xbc[(size_t)(b * T + t0 + s) * CONVCH + hh * HEADDIM + p];
            sUt[p][s]   = f2bf(u);
            sUtld[p][s] = f2bf(u * __expf(Lc_last - sLc[s]));
        }
        __syncthreads();

        // G = C·B^T, masked decay -> sM (4 tiles on waves 0..3)
        if (wave < 4) {
            const int ti = wave >> 1, ts = wave & 1;
            v8f acc = {};
            const unsigned short* arow = &sC[ti * 16 + l15][0];
            const unsigned short* wrow = &sB[ts * 16 + l15][0];
            for (int k0 = 0; k0 < DSTATE; k0 += 32)
                acc = wmma_bf16(load_a_frag(arow, k0, half),
                                load_b_frag(wrow, k0, half), acc);
            const int i0 = ti * 16 + half * 8, scol = ts * 16 + l15;
#pragma unroll
            for (int r = 0; r < 8; ++r) {
                int i = i0 + r;
                float v = (scol <= i) ? acc[r] * __expf(sLc[i] - sLc[scol]) : 0.0f;
                sM[i][scol] = f2bf(v);
            }
        }
        __syncthreads();

        // Y = M·U + exp(Lc_i) * (C·S^T)  (8 tiles: 2i x 4p)
        {
            const int ti = wave >> 2, tp = wave & 3;
            v8f acc1 = {};
            acc1 = wmma_bf16(load_a_frag(&sM[ti * 16 + l15][0], 0, half),
                             load_b_frag(&sUt[tp * 16 + l15][0], 0, half), acc1);
            v8f acc2 = {};
            const unsigned short* arow = &sC[ti * 16 + l15][0];
            const unsigned short* wrow = &sS[tp * 16 + l15][0];
            for (int k0 = 0; k0 < DSTATE; k0 += 32)
                acc2 = wmma_bf16(load_a_frag(arow, k0, half),
                                 load_b_frag(wrow, k0, half), acc2);
            const int i0 = ti * 16 + half * 8, pcol = tp * 16 + l15;
#pragma unroll
            for (int r = 0; r < 8; ++r) {
                int i = i0 + r;
                float y = acc1[r] + __expf(sLc[i]) * acc2[r];
                yscan[(size_t)(b * T + t0 + i) * DINNER + hh * HEADDIM + pcol] = y;
            }
        }
        __syncthreads();

        // S = exp(Lc_last)*S + Utld·B  (32 tiles: 4p x 8n, 4 per wave)
        float olds[4][8];
#pragma unroll
        for (int j = 0; j < 4; ++j) {
            const int idx = wave * 4 + j;
            const int p0 = (idx >> 3) * 16 + half * 8;
            const int n  = (idx & 7) * 16 + l15;
#pragma unroll
            for (int r = 0; r < 8; ++r) olds[j][r] = bf2f(sS[p0 + r][n]);
        }
        __syncthreads();
        const float Pl = __expf(Lc_last);
#pragma unroll
        for (int j = 0; j < 4; ++j) {
            const int idx = wave * 4 + j;
            const int tp = idx >> 3, tn = idx & 7;
            v8f acc = {};
            acc = wmma_bf16(load_a_frag(&sUtld[tp * 16 + l15][0], 0, half),
                            load_b_frag(&sBT[tn * 16 + l15][0], 0, half), acc);
            const int p0 = tp * 16 + half * 8, n = tn * 16 + l15;
#pragma unroll
            for (int r = 0; r < 8; ++r)
                sS[p0 + r][n] = f2bf(Pl * olds[j][r] + acc[r]);
        }
        __syncthreads();
    }
}

// ---------------- 7) y = (yscan + D*xs) * silu(z); RMSNorm * gw -> bf16 ----------------
__global__ __launch_bounds__(256) void gate_norm_kernel(
    const float* __restrict__ yscan, const float* __restrict__ xbc,
    const float* __restrict__ zx, const float* __restrict__ Dd,
    const float* __restrict__ gw, unsigned short* __restrict__ outbf)
{
    __shared__ float red[256];
    const int row = blockIdx.x;                       // 0..BT
    const float* yr = yscan + (size_t)row * DINNER;
    const float* xr = xbc   + (size_t)row * CONVCH;
    const float* zr = zx    + (size_t)row * DINPROJ;
    float vals[8];
    float ss = 0.0f;
#pragma unroll
    for (int j = 0; j < 8; ++j) {
        int i = threadIdx.x + j * 256;
        float y = yr[i] + Dd[i >> 6] * xr[i];
        y *= silu(zr[i]);
        vals[j] = y;
        ss += y * y;
    }
    red[threadIdx.x] = ss;
    __syncthreads();
    for (int st = 128; st > 0; st >>= 1) {
        if (threadIdx.x < st) red[threadIdx.x] += red[threadIdx.x + st];
        __syncthreads();
    }
    float scale = rsqrtf(red[0] / DINNER + EPS);
#pragma unroll
    for (int j = 0; j < 8; ++j) {
        int i = threadIdx.x + j * 256;
        outbf[(size_t)row * DINNER + i] = f2bf(vals[j] * scale * gw[i]);
    }
}

// ---------------- 8) final RMSNorm over DIM -> bf16 ----------------
__global__ __launch_bounds__(256) void final_norm_kernel(
    const float* __restrict__ h, const float* __restrict__ gamma,
    unsigned short* __restrict__ outbf)
{
    __shared__ float red[256];
    const int row = blockIdx.x;
    const float* hr = h + (size_t)row * DIM;
    float vals[4];
    float ss = 0.0f;
#pragma unroll
    for (int j = 0; j < 4; ++j) {
        vals[j] = hr[threadIdx.x + j * 256];
        ss += vals[j] * vals[j];
    }
    red[threadIdx.x] = ss;
    __syncthreads();
    for (int st = 128; st > 0; st >>= 1) {
        if (threadIdx.x < st) red[threadIdx.x] += red[threadIdx.x + st];
        __syncthreads();
    }
    float scale = rsqrtf(red[0] / DIM + EPS);
#pragma unroll
    for (int j = 0; j < 4; ++j) {
        int i = threadIdx.x + j * 256;
        outbf[(size_t)row * DIM + i] = f2bf(vals[j] * scale * gamma[i]);
    }
}

// ---------------- workspace layout (bytes) ----------------
constexpr size_t OFF_H     = 0;                                   // BT*DIM f32      8 MB
constexpr size_t OFF_ABF   = OFF_H     + (size_t)BT * DIM * 4;    // BT*DINNER bf16  8 MB
constexpr size_t OFF_ZX    = OFF_ABF   + (size_t)BT * DINNER * 2; // BT*DINPROJ f32 36 MB
constexpr size_t OFF_XBC   = OFF_ZX    + (size_t)BT * DINPROJ * 4;// BT*CONVCH f32  19 MB
constexpr size_t OFF_DT    = OFF_XBC   + (size_t)BT * CONVCH * 4;
constexpr size_t OFF_DA    = OFF_DT    + (size_t)BT * NHEADS * 4;
constexpr size_t OFF_YS    = OFF_DA    + (size_t)BT * NHEADS * 4; // BT*DINNER f32  16 MB
constexpr size_t OFF_WBF   = OFF_YS    + (size_t)BT * DINNER * 4; // max weight bf16 9 MB
constexpr size_t OFF_EMBBF = OFF_WBF   + (size_t)DINPROJ * DIM * 2;
// total ~ 98.4 MB

extern "C" void kernel_launch(void* const* d_in, const int* in_sizes, int n_in,
                              void* d_out, int out_size, void* d_ws, size_t ws_size,
                              hipStream_t stream)
{
    const int*   x     = (const int*)  d_in[0];
    const float* embed = (const float*)d_in[1];
    const float* inW   = (const float*)d_in[2];
    const float* convW = (const float*)d_in[3];
    const float* convB = (const float*)d_in[4];
    const float* dtb   = (const float*)d_in[5];
    const float* alog  = (const float*)d_in[6];
    const float* Dd    = (const float*)d_in[7];
    const float* gw    = (const float*)d_in[8];
    const float* outW  = (const float*)d_in[9];
    const float* fnw   = (const float*)d_in[10];
    float* out = (float*)d_out;

    char* ws = (char*)d_ws;
    float*          h     = (float*)(ws + OFF_H);
    unsigned short* abf   = (unsigned short*)(ws + OFF_ABF);
    float*          zx    = (float*)(ws + OFF_ZX);
    float*          xbc   = (float*)(ws + OFF_XBC);
    float*          dtbuf = (float*)(ws + OFF_DT);
    float*          dabuf = (float*)(ws + OFF_DA);
    float*          ys    = (float*)(ws + OFF_YS);
    unsigned short* wbf   = (unsigned short*)(ws + OFF_WBF);
    unsigned short* embbf = (unsigned short*)(ws + OFF_EMBBF);

    auto gemm_blocks = [](int M, int N) {
        int tiles_n = N / 16;
        int ngroups = (tiles_n + GEMM_NT - 1) / GEMM_NT;
        return (M / 64) * ngroups;
    };

    // 1) h = embed[x]; embed -> bf16 for final logits GEMM
    embed_gather_kernel<<<(BT * DIM) / 256, 256, 0, stream>>>(x, embed, h);
    cvt_f32_bf16_kernel<<<1024, 256, 0, stream>>>(embed, embbf, (size_t)VOCAB * DIM);

    for (int l = 0; l < DEPTH; ++l) {
        // 2) A = bf16(h); W = bf16(in_proj_w[l])
        cvt_f32_bf16_kernel<<<2048, 256, 0, stream>>>(h, abf, (size_t)BT * DIM);
        cvt_f32_bf16_kernel<<<4096, 256, 0, stream>>>(
            inW + (size_t)l * DINPROJ * DIM, wbf, (size_t)DINPROJ * DIM);
        // 3) zxbcdt = h @ Wp^T   (M=2048, N=4384, K=1024)
        gemm_bf16_wmma_kernel<<<gemm_blocks(BT, DINPROJ), 256, 0, stream>>>(
            abf, wbf, nullptr, zx, BT, DINPROJ, DIM);
        // 4) conv + silu
        conv_silu_kernel<<<((size_t)BT * CONVCH) / 256, 256, 0, stream>>>(
            zx, convW + (size_t)l * CONVCH * DCONV, convB + (size_t)l * CONVCH, xbc);
        // 5) dt / dA
        dt_act_kernel<<<(BT * NHEADS) / 256, 256, 0, stream>>>(
            zx, dtb + l * NHEADS, alog + l * NHEADS, dtbuf, dabuf);
        // 6) chunked WMMA scan (64 blocks, 32 sequential chunks each)
        chunk_scan_kernel<<<B * NHEADS, 256, 0, stream>>>(xbc, dtbuf, dabuf, ys);
        // 7) gate + RMSNorm -> bf16 A for out_proj
        gate_norm_kernel<<<BT, 256, 0, stream>>>(
            ys, xbc, zx, Dd + l * NHEADS, gw + (size_t)l * DINNER, abf);
        // 8) W = bf16(out_proj_w[l]); h = ynorm @ Wo^T + h (residual in epilogue)
        cvt_f32_bf16_kernel<<<4096, 256, 0, stream>>>(
            outW + (size_t)l * DIM * DINNER, wbf, (size_t)DIM * DINNER);
        gemm_bf16_wmma_kernel<<<gemm_blocks(BT, DIM), 256, 0, stream>>>(
            abf, wbf, h, h, BT, DIM, DINNER);
    }

    // 9) final RMSNorm -> bf16; logits = hn @ embed^T  (M=2048, N=256, K=1024)
    final_norm_kernel<<<BT, 256, 0, stream>>>(h, fnw, abf);
    gemm_bf16_wmma_kernel<<<gemm_blocks(BT, VOCAB), 256, 0, stream>>>(
        abf, embbf, nullptr, out, BT, VOCAB, DIM);
}
